// LocalWindowAttention_1254130450883
// MI455X (gfx1250) — matrix operand order repaired and checked by reading, not compiled, not run on hardware
//
#include <hip/hip_runtime.h>
#include <hip/hip_bf16.h>

// ---------------------------------------------------------------------------
// Local window (block-causal) attention for MI455X / gfx1250, wave32 + WMMA.
//   B=1, T=2048, D=1024, H=16, HD=64, W=128
// Pipeline (f16 compute, f32 accumulate):
//   1) cast x -> f16; cast+transpose Wq/Wk/Wv/Wo -> [N][K] f16
//   2) LDS-tiled WMMA GEMM, double-buffered via GLOBAL_LOAD_ASYNC_TO_LDS
//      (ASYNCcnt-tracked, global->LDS with no VGPR round trip):
//        Qh=[T,D] f16, Kh=[T,D] f16, Vt=[H][HD][T] f16 (pre-transposed)
//   3) flash attention, 32-key chunks (block-causal mask is chunk-aligned so
//      no masking needed); K/V chunks async-staged in LDS shared by 8 waves
//   4) LDS-tiled WMMA GEMM: out = Oc @ Wo  (f32 output)
// ---------------------------------------------------------------------------

typedef __attribute__((ext_vector_type(16))) _Float16 v16h;
typedef __attribute__((ext_vector_type(8)))  _Float16 v8h;
typedef __attribute__((ext_vector_type(8)))  float    v8f;

#define LOG2E 1.44269504088896340736f

static constexpr int Tn = 2048;
static constexpr int Dn = 1024;
static constexpr int Hn = 16;
static constexpr int HDn = 64;
static constexpr int Wn = 128;

// ---- CDNA5 async global->LDS copy (CDNA5 ISA) --------
// 16 bytes per lane, tracked by ASYNCcnt. lds_off = low 32 bits of the flat
// LDS address (aperture maps addr[31:0] to the LDS offset).
__device__ __forceinline__ void async_copy16(unsigned lds_off, const _Float16* g) {
  asm volatile("global_load_async_to_lds_b128 %0, %1, off"
               :: "v"(lds_off), "v"(g) : "memory");
}
__device__ __forceinline__ void wait_async_all() {
  asm volatile("s_wait_asynccnt 0x0" ::: "memory");
}
__device__ __forceinline__ unsigned lds_addr32(const void* p) {
  return (unsigned)(unsigned long long)p;
}

// ---- WMMA fragment loaders (layouts from CDNA5 ISA §7.12.2) -----

// A-matrix 16x32 f16: lane holds row m=lane%16; halves 0..7 = K[8*hi .. +7],
// halves 8..15 = K[16+8*hi .. +7]  (hi = lane/16).
__device__ __forceinline__ v16h load_a16(const _Float16* base, int ld, int row,
                                         int hi, int k0) {
  union { v16h v; v8h h[2]; } u;
  const _Float16* p = base + (size_t)row * ld + k0 + 8 * hi;
  u.h[0] = *(const v8h*)(p);
  u.h[1] = *(const v8h*)(p + 16);
  return u.v;
}

// B-matrix 32x16 f16 from a transposed ([N][K] row-major) buffer:
// lane holds column n=lane%16; halves 0..7 = K[k0+8*hi .. +7],
// halves 8..15 = K[k0+16+8*hi .. +7].
__device__ __forceinline__ v16h load_bT(const _Float16* bt, int ld, int col,
                                        int hi, int k0) {
  union { v16h v; v8h h[2]; } u;
  const _Float16* p = bt + (size_t)col * ld + k0 + 8 * hi;
  u.h[0] = *(const v8h*)(p);
  u.h[1] = *(const v8h*)(p + 16);
  return u.v;
}

// ---- conversion kernels ----------------------------------------------------

__global__ void __launch_bounds__(256)
cast_f32_to_f16(const float* __restrict__ in, _Float16* __restrict__ out, int n) {
  int i = blockIdx.x * blockDim.x + threadIdx.x;
  if (i < n) out[i] = (_Float16)in[i];
}

// wt[r][c] = w[c][r]  (both n x n, n = 1024)
__global__ void __launch_bounds__(256)
transpose_cast(const float* __restrict__ w, _Float16* __restrict__ wt, int n) {
  int i = blockIdx.x * blockDim.x + threadIdx.x;
  if (i < n * n) {
    int r = i / n, c = i % n;
    wt[i] = (_Float16)w[(size_t)c * n + r];
  }
}

// ---- LDS-tiled async-double-buffered WMMA GEMM -----------------------------
// C[M,N] = A[M,K] (f16 row-major) x B (given as Bt = [N][K] f16 row-major)
// Block tile: 128M x 128N, k-step 32. 8 waves; wave w computes rows
// [bm*128 + w*16, +16) x all 128 cols (8 accumulators).
// mode 0: f16 out [M,N] | mode 1: f16 Vt[h][hd][t] | mode 2: f32 out [M,N]
__global__ void __launch_bounds__(256)
gemm_wmma(const _Float16* __restrict__ A, const _Float16* __restrict__ Bt,
          void* __restrict__ out, int M, int N, int K, int mode) {
  __shared__ _Float16 As[2][128][32];   // 8 KB x2
  __shared__ _Float16 Bs[2][128][32];   // 8 KB x2

  const int t = threadIdx.x;
  const int lane = t & 31, wave = t >> 5;
  const int nbn = N >> 7;
  const int bm = (int)blockIdx.x / nbn;
  const int bn = (int)blockIdx.x % nbn;
  const int m = lane & 15, hi = lane >> 4;

  // stagers: each of 256 threads moves 32 B of A and 32 B of B per k-step
  const int lrow = t >> 1;             // 0..127
  const int lseg = (t & 1) * 16;       // halves 0..15 or 16..31
  const _Float16* gA = A  + (size_t)(bm * 128 + lrow) * K + lseg;
  const _Float16* gB = Bt + (size_t)(bn * 128 + lrow) * K + lseg;
  const unsigned ldsA0 = lds_addr32(&As[0][lrow][lseg]);
  const unsigned ldsB0 = lds_addr32(&Bs[0][lrow][lseg]);
  const unsigned bufStride = 128u * 32u * 2u;   // 8192 B per stage

  // prologue: async-stage k-step 0 into buffer 0 (no VGPR round trip)
  async_copy16(ldsA0,      gA);
  async_copy16(ldsA0 + 16, gA + 8);
  async_copy16(ldsB0,      gB);
  async_copy16(ldsB0 + 16, gB + 8);

  v8f acc[8] = {};
  const int nk = K >> 5;
  for (int kt = 0; kt < nk; ++kt) {
    const int buf = kt & 1;
    wait_async_all();                  // our async fills of stage[buf] landed
    __syncthreads();                   // ... and everyone else's too
    if (kt + 1 < nk) {                 // async-fill the other stage; readers
      const unsigned nb = (unsigned)(buf ^ 1) * bufStride;  // of it finished
      const _Float16* pa = gA + (size_t)(kt + 1) * 32;      // before the sync
      const _Float16* pb = gB + (size_t)(kt + 1) * 32;
      async_copy16(ldsA0 + nb,      pa);
      async_copy16(ldsA0 + nb + 16, pa + 8);
      async_copy16(ldsB0 + nb,      pb);
      async_copy16(ldsB0 + nb + 16, pb + 8);
      if (kt + 2 < nk) {               // gfx1250 global_prefetch_b8 hint
        __builtin_prefetch(gA + (size_t)(kt + 2) * 32, 0, 3);
        __builtin_prefetch(gB + (size_t)(kt + 2) * 32, 0, 3);
      }
    }
    // Issue ALL 18 ds_loads into distinct regs, fence the scheduler, then run
    // the WMMA burst: in-order LDS returns let the backend use partial
    // s_wait_dscnt so later loads complete underneath earlier WMMAs.
    v16h a = load_a16(&As[buf][0][0], 32, wave * 16 + m, hi, 0);
    v16h bfr[8];
#pragma unroll
    for (int j = 0; j < 8; ++j)
      bfr[j] = load_bT(&Bs[buf][0][0], 32, 16 * j + m, hi, 0);
    __builtin_amdgcn_sched_barrier(0);
#pragma unroll
    for (int j = 0; j < 8; ++j)
      acc[j] = __builtin_amdgcn_wmma_f32_16x16x32_f16(
          false, a, false, bfr[j], (short)0, acc[j], false, false);
  }

#pragma unroll
  for (int j = 0; j < 8; ++j) {
#pragma unroll
    for (int r = 0; r < 8; ++r) {
      const int rr = bm * 128 + wave * 16 + r + 8 * hi;  // C/D: lane=col, vgpr=row
      const int cc = bn * 128 + 16 * j + m;
      const float v = acc[j][r];
      if (mode == 2) {
        ((float*)out)[(size_t)rr * N + cc] = v;
      } else if (mode == 0) {
        ((_Float16*)out)[(size_t)rr * N + cc] = (_Float16)v;
      } else {  // V transposed per head: Vt[h][hd][t]
        const int h = cc >> 6, hd = cc & 63;
        ((_Float16*)out)[((size_t)h * HDn + hd) * Tn + rr] = (_Float16)v;
      }
    }
  }
}

// ---- flash attention, block-causal (kend = (blk+1)*128, chunk-aligned) -----
// One block = one head x one 128-row query window. All 8 waves share the same
// key chunks, so K/V chunks are async-staged once per block (double buffered).
__global__ void __launch_bounds__(256)
attn_wmma(const _Float16* __restrict__ Qh,   // [T, D] f16
          const _Float16* __restrict__ Kh,   // [T, D] f16
          const _Float16* __restrict__ Vt,   // [H][HD][T] f16
          _Float16* __restrict__ Oc) {       // [T, D] f16
  __shared__ _Float16 Ks[2][32][64];         // 4 KB x2: 32 keys x 64 hd
  __shared__ _Float16 Vs[2][64][32];         // 4 KB x2: 64 hd x 32 keys
  __shared__ _Float16 pbuf[8][16][32];       // per-wave P staging (D->A layout)

  const int t = threadIdx.x;
  const int lane = t & 31, wave = t >> 5;
  const int h = blockIdx.x & (Hn - 1);
  const int blk = blockIdx.x >> 4;           // 128-row query window
  const int q0 = blk * Wn + wave * 16;
  const int m = lane & 15, hi = lane >> 4;

  // stagers: 256 threads x 16 B cover each 4 KB chunk
  const int krow = t >> 3, koff = (t & 7) * 8;   // Ks[32][64]
  const int vrow = t >> 2, voff = (t & 3) * 8;   // Vs[64][32]
  const _Float16* gK = Kh + h * HDn + (size_t)krow * Dn + koff;              // +kc*D
  const _Float16* gV = Vt + (size_t)h * HDn * Tn + (size_t)vrow * Tn + voff; // +kc
  const unsigned ldsK0 = lds_addr32(&Ks[0][krow][koff]);
  const unsigned ldsV0 = lds_addr32(&Vs[0][vrow][voff]);
  const unsigned kvStride = 32u * 64u * 2u;      // 4096 B per stage

  // Q fragments for the two HD chunks (0..31, 32..63), reused every iteration
  const v16h qa0 = load_a16(Qh + h * HDn, Dn, q0 + m, hi, 0);
  const v16h qa1 = load_a16(Qh + h * HDn, Dn, q0 + m, hi, 32);

  // prologue: async-stage key chunk 0 into buffer 0
  async_copy16(ldsK0, gK);
  async_copy16(ldsV0, gV);

  float m_run[8], l_run[8];
  v8f o[4] = {};
#pragma unroll
  for (int r = 0; r < 8; ++r) { m_run[r] = -1e30f; l_run[r] = 0.0f; }

  const int kend = (blk + 1) * Wn;   // visible keys: [0, kend), multiple of 128
  const int nit = kend >> 5;
  for (int it = 0; it < nit; ++it) {
    const int buf = it & 1;
    const int kc = it * 32;
    wait_async_all();
    __syncthreads();                 // K/V stage[buf] visible to all waves
    if (it + 1 < nit) {              // async-fill the other stage
      const unsigned nb = (unsigned)(buf ^ 1) * kvStride;
      async_copy16(ldsK0 + nb, gK + (size_t)(kc + 32) * Dn);
      async_copy16(ldsV0 + nb, gV + kc + 32);
    }

    // S tile: 16 queries x 32 keys, accumulate over HD=64 (two k-chunks).
    // Preload all four K fragments, fence, then the WMMA burst.
    v8f s0 = {}, s1 = {};
    {
      v16h b00 = load_bT(&Ks[buf][0][0], 64, m, hi, 0);
      v16h b01 = load_bT(&Ks[buf][0][0], 64, m, hi, 32);
      v16h b10 = load_bT(&Ks[buf][0][0], 64, 16 + m, hi, 0);
      v16h b11 = load_bT(&Ks[buf][0][0], 64, 16 + m, hi, 32);
      __builtin_amdgcn_sched_barrier(0);
      s0 = __builtin_amdgcn_wmma_f32_16x16x32_f16(false, qa0, false, b00, (short)0, s0, false, false);
      s0 = __builtin_amdgcn_wmma_f32_16x16x32_f16(false, qa1, false, b01, (short)0, s0, false, false);
      s1 = __builtin_amdgcn_wmma_f32_16x16x32_f16(false, qa0, false, b10, (short)0, s1, false, false);
      s1 = __builtin_amdgcn_wmma_f32_16x16x32_f16(false, qa1, false, b11, (short)0, s1, false, false);
    }

    // Online softmax; row r+8*hi lives across the lane's 16-lane half.
    const float scale = 0.125f;  // 1/sqrt(HD)
    float p0[8], p1[8], alpha[8];
#pragma unroll
    for (int r = 0; r < 8; ++r) {
      float a0 = s0[r] * scale, a1 = s1[r] * scale;
      float mx = fmaxf(a0, a1);
      mx = fmaxf(mx, __shfl_xor(mx, 1, 32));
      mx = fmaxf(mx, __shfl_xor(mx, 2, 32));
      mx = fmaxf(mx, __shfl_xor(mx, 4, 32));
      mx = fmaxf(mx, __shfl_xor(mx, 8, 32));
      const float mn = fmaxf(m_run[r], mx);
      const float al = exp2f((m_run[r] - mn) * LOG2E);
      const float e0 = exp2f((a0 - mn) * LOG2E);
      const float e1 = exp2f((a1 - mn) * LOG2E);
      float rs = e0 + e1;
      rs += __shfl_xor(rs, 1, 32);
      rs += __shfl_xor(rs, 2, 32);
      rs += __shfl_xor(rs, 4, 32);
      rs += __shfl_xor(rs, 8, 32);
      l_run[r] = l_run[r] * al + rs;
      m_run[r] = mn;
      alpha[r] = al; p0[r] = e0; p1[r] = e1;
    }
#pragma unroll
    for (int j = 0; j < 4; ++j)
#pragma unroll
      for (int r = 0; r < 8; ++r) o[j][r] *= alpha[r];

    // Redistribute P from C/D layout to A layout via per-wave LDS tile.
#pragma unroll
    for (int r = 0; r < 8; ++r) {
      const int rowp = r + 8 * hi;
      pbuf[wave][rowp][m]      = (_Float16)p0[r];
      pbuf[wave][rowp][16 + m] = (_Float16)p1[r];
    }
    __syncthreads();   // uniform across waves; also orders pbuf store->load
    v16h pa;
    {
      union { v16h v; v8h hh[2]; } u;
      const _Float16* p = &pbuf[wave][m][8 * hi];
      u.hh[0] = *(const v8h*)(p);
      u.hh[1] = *(const v8h*)(p + 16);
      pa = u.v;
    }

    // O += P x V  (B from LDS Vs: col = hd, k = local key); preload + fence
    v16h vb[4];
#pragma unroll
    for (int j = 0; j < 4; ++j)
      vb[j] = load_bT(&Vs[buf][0][0], 32, 16 * j + m, hi, 0);
    __builtin_amdgcn_sched_barrier(0);
#pragma unroll
    for (int j = 0; j < 4; ++j)
      o[j] = __builtin_amdgcn_wmma_f32_16x16x32_f16(
          false, pa, false, vb[j], (short)0, o[j], false, false);
  }

  // normalize + store f16 [T, D]
#pragma unroll
  for (int r = 0; r < 8; ++r) {
    const float inv = 1.0f / l_run[r];
    const int rr = q0 + r + 8 * hi;
#pragma unroll
    for (int j = 0; j < 4; ++j)
      Oc[(size_t)rr * Dn + h * HDn + 16 * j + m] = (_Float16)(o[j][r] * inv);
  }
}

// ---------------------------------------------------------------------------

extern "C" void kernel_launch(void* const* d_in, const int* in_sizes, int n_in,
                              void* d_out, int out_size, void* d_ws, size_t ws_size,
                              hipStream_t stream) {
  const float* x  = (const float*)d_in[0];
  const float* Wq = (const float*)d_in[1];
  const float* Wk = (const float*)d_in[2];
  const float* Wv = (const float*)d_in[3];
  const float* Wo = (const float*)d_in[4];

  // Workspace carve-up (f16): 28 MB total
  _Float16* ws  = (_Float16*)d_ws;
  _Float16* xh  = ws;                           // T*D
  _Float16* Wqt = xh  + (size_t)Tn * Dn;        // D*D each, [N][K]
  _Float16* Wkt = Wqt + (size_t)Dn * Dn;
  _Float16* Wvt = Wkt + (size_t)Dn * Dn;
  _Float16* Wot = Wvt + (size_t)Dn * Dn;
  _Float16* Qh  = Wot + (size_t)Dn * Dn;        // [T, D]
  _Float16* Kh  = Qh  + (size_t)Tn * Dn;        // [T, D]
  _Float16* Vt  = Kh  + (size_t)Tn * Dn;        // [H][HD][T]
  _Float16* Oc  = Vt  + (size_t)Tn * Dn;        // [T, D]

  // 1) precision conversion + weight transposes
  {
    const int n = Tn * Dn;
    cast_f32_to_f16<<<(n + 255) / 256, 256, 0, stream>>>(x, xh, n);
    const int nw = Dn * Dn;
    transpose_cast<<<(nw + 255) / 256, 256, 0, stream>>>(Wq, Wqt, Dn);
    transpose_cast<<<(nw + 255) / 256, 256, 0, stream>>>(Wk, Wkt, Dn);
    transpose_cast<<<(nw + 255) / 256, 256, 0, stream>>>(Wv, Wvt, Dn);
    transpose_cast<<<(nw + 255) / 256, 256, 0, stream>>>(Wo, Wot, Dn);
  }

  // 2) QKV projections: (T/128)*(D/128) = 128 blocks x 256 threads
  const int gemm_blocks = (Tn / 128) * (Dn / 128);
  gemm_wmma<<<gemm_blocks, 256, 0, stream>>>(xh, Wqt, Qh, Tn, Dn, Dn, 0);
  gemm_wmma<<<gemm_blocks, 256, 0, stream>>>(xh, Wkt, Kh, Tn, Dn, Dn, 0);
  gemm_wmma<<<gemm_blocks, 256, 0, stream>>>(xh, Wvt, Vt, Tn, Dn, Dn, 1);

  // 3) attention: one block = one head x one 128-row window (16*16 = 256)
  attn_wmma<<<Hn * (Tn / Wn), 256, 0, stream>>>(Qh, Kh, Vt, Oc);

  // 4) output projection -> f32
  gemm_wmma<<<gemm_blocks, 256, 0, stream>>>(Oc, Wot, (float*)d_out, Tn, Dn, Dn, 2);
}
